// LearningModel_87557203296995
// MI455X (gfx1250) — matrix operand order repaired and checked by reading, not compiled, not run on hardware
//
#include <hip/hip_runtime.h>
#include <math.h>

// ---------------- problem constants ----------------
#define E        256
#define RULES    8
#define STEPS    512
#define NPS      256          // nodes per step
#define NINIT    1024
#define TOTAL    (NINIT + STEPS * NPS)   // 132096
#define MMASK    65536

// LDS row strides (elements), keep 16B divisibility for b128 loads
#define PSTR 520              // 512 + 8 pad  (1040 B, /16 ok)
#define HSTR 264              // 256 + 8 pad  (528 B, /16 ok)

#define SCAN_BLOCKS 8
#define EVAL_BLOCKS (MMASK / 32)   // 2048

typedef __attribute__((ext_vector_type(16))) __bf16 v16bf;
typedef __attribute__((ext_vector_type(8)))  __bf16 v8bf;
typedef __attribute__((ext_vector_type(8)))  float  v8f;

union ABf { v16bf v; v8bf h[2]; };

__device__ __forceinline__ unsigned short f2bf(float f) {
  union { float f; unsigned u; } v; v.f = f;
  unsigned r = v.u + 0x7FFFu + ((v.u >> 16) & 1u);   // RNE
  return (unsigned short)(r >> 16);
}

__device__ __forceinline__ float softplusf(float x) {
  return x > 0.f ? x + log1pf(expf(-x)) : log1pf(expf(x));
}

// ---------------- prep: bf16 weight transposes + init copy + barrier reset ----
__global__ __launch_bounds__(256) void prep_kernel(
    const float* __restrict__ init_vecs,
    const float* __restrict__ W1, const float* __restrict__ W2,
    const float* __restrict__ eW1,
    float* __restrict__ vectors,
    unsigned short* __restrict__ W1T, unsigned short* __restrict__ W2T,
    unsigned short* __restrict__ eW1T,
    unsigned int* __restrict__ bar)
{
  size_t idx = (size_t)blockIdx.x * 256 + threadIdx.x;
  if (idx == 0) *bar = 0u;
  const size_t n1 = (size_t)RULES * 256 * 512;   // 1048576  W1T[r][j][k]
  const size_t n2 = (size_t)RULES * 256 * 256;   // 524288   W2T[r][j][k]
  const size_t n3 = 256 * 256;                   // 65536    eW1T[j][k]
  const size_t n4 = (size_t)NINIT * E;           // 262144   init copy
  if (idx < n1) {
    size_t r = idx >> 17, j = (idx >> 9) & 255, k = idx & 511;
    W1T[idx] = f2bf(W1[(r * 512 + k) * 256 + j]);
  } else if (idx < n1 + n2) {
    size_t t = idx - n1;
    size_t r = t >> 16, j = (t >> 8) & 255, k = t & 255;
    W2T[t] = f2bf(W2[(r * 256 + k) * 256 + j]);
  } else if (idx < n1 + n2 + n3) {
    size_t t = idx - n1 - n2;
    size_t j = t >> 8, k = t & 255;
    eW1T[t] = f2bf(eW1[k * 256 + j]);
  } else if (idx < n1 + n2 + n3 + n4) {
    size_t t = idx - n1 - n2 - n3;
    vectors[t] = init_vecs[t];
  }
}

// ---------------- persistent scan kernel: 8 WGs x 256 threads ---------------
__global__ __launch_bounds__(256) void scan_kernel(
    float* __restrict__ vectors,
    const unsigned short* __restrict__ W1T,
    const unsigned short* __restrict__ W2T,
    const float* __restrict__ b1, const float* __restrict__ b2,
    const int* __restrict__ pars,     // [STEPS*NPS*2]
    const int* __restrict__ rules,    // [STEPS]
    unsigned int* __restrict__ bar)
{
  __shared__ unsigned short Plds[32 * PSTR];
  __shared__ unsigned short Hlds[32 * HSTR];

  const int tid  = threadIdx.x;
  const int lane = tid & 31;
  const int wave = tid >> 5;
  const int hi   = lane >> 4;         // half-wave id
  const int l15  = lane & 15;
  const int rt   = wave & 1;          // row tile 0/1 (16 rows each)
  const int q    = wave >> 1;         // 0..3 -> col tiles q + 4j
  const int i0   = blockIdx.x * 32;   // this WG's 32-row stripe

  for (int s = 0; s < STEPS; ++s) {
    const int r = rules[s];

    // ---- phase 0: gather parents -> bf16 LDS [32][512]
    for (int idx = tid; idx < 32 * 512; idx += 256) {
      int row = idx >> 9, k = idx & 511;
      int par = pars[((s * NPS) + i0 + row) * 2 + (k >> 8)];
      Plds[row * PSTR + k] = f2bf(vectors[(size_t)par * E + (k & 255)]);
    }
    __syncthreads();

    // ---- phase 1: H = relu(P @ W1[r] + b1[r])
    {
      v8f acc0 = {}, acc1 = {}, acc2 = {}, acc3 = {};
      const int arow = (rt * 16 + l15) * PSTR;
      const unsigned short* w1r = W1T + (size_t)r * 256 * 512;
      #pragma unroll 4
      for (int ks = 0; ks < 16; ++ks) {
        const int k0 = ks * 32;
        ABf a;
        a.h[0] = *(const v8bf*)&Plds[arow + k0 + hi * 16];
        a.h[1] = *(const v8bf*)&Plds[arow + k0 + hi * 16 + 8];
        const int cb = q * 16 + l15;
        v16bf bB0 = *(const v16bf*)&w1r[(size_t)(cb)       * 512 + k0 + hi * 16];
        v16bf bB1 = *(const v16bf*)&w1r[(size_t)(cb +  64) * 512 + k0 + hi * 16];
        v16bf bB2 = *(const v16bf*)&w1r[(size_t)(cb + 128) * 512 + k0 + hi * 16];
        v16bf bB3 = *(const v16bf*)&w1r[(size_t)(cb + 192) * 512 + k0 + hi * 16];
        acc0 = __builtin_amdgcn_wmma_f32_16x16x32_bf16(false, a.v, false, bB0, (short)0, acc0, false, false);
        acc1 = __builtin_amdgcn_wmma_f32_16x16x32_bf16(false, a.v, false, bB1, (short)0, acc1, false, false);
        acc2 = __builtin_amdgcn_wmma_f32_16x16x32_bf16(false, a.v, false, bB2, (short)0, acc2, false, false);
        acc3 = __builtin_amdgcn_wmma_f32_16x16x32_bf16(false, a.v, false, bB3, (short)0, acc3, false, false);
      }
      v8f* accs[4] = { &acc0, &acc1, &acc2, &acc3 };
      #pragma unroll
      for (int j = 0; j < 4; ++j) {
        const int col = (q + 4 * j) * 16 + l15;
        const float bias = b1[r * 256 + col];
        #pragma unroll
        for (int v = 0; v < 8; ++v) {
          float hv = (*accs[j])[v] + bias;
          hv = hv > 0.f ? hv : 0.f;
          Hlds[(rt * 16 + v + hi * 8) * HSTR + col] = f2bf(hv);
        }
      }
    }
    __syncthreads();

    // ---- phase 2: OUT = H @ W2[r] + b2[r] -> vectors rows
    {
      v8f acc0 = {}, acc1 = {}, acc2 = {}, acc3 = {};
      const int arow = (rt * 16 + l15) * HSTR;
      const unsigned short* w2r = W2T + (size_t)r * 256 * 256;
      #pragma unroll
      for (int ks = 0; ks < 8; ++ks) {
        const int k0 = ks * 32;
        ABf a;
        a.h[0] = *(const v8bf*)&Hlds[arow + k0 + hi * 16];
        a.h[1] = *(const v8bf*)&Hlds[arow + k0 + hi * 16 + 8];
        const int cb = q * 16 + l15;
        v16bf bB0 = *(const v16bf*)&w2r[(size_t)(cb)       * 256 + k0 + hi * 16];
        v16bf bB1 = *(const v16bf*)&w2r[(size_t)(cb +  64) * 256 + k0 + hi * 16];
        v16bf bB2 = *(const v16bf*)&w2r[(size_t)(cb + 128) * 256 + k0 + hi * 16];
        v16bf bB3 = *(const v16bf*)&w2r[(size_t)(cb + 192) * 256 + k0 + hi * 16];
        acc0 = __builtin_amdgcn_wmma_f32_16x16x32_bf16(false, a.v, false, bB0, (short)0, acc0, false, false);
        acc1 = __builtin_amdgcn_wmma_f32_16x16x32_bf16(false, a.v, false, bB1, (short)0, acc1, false, false);
        acc2 = __builtin_amdgcn_wmma_f32_16x16x32_bf16(false, a.v, false, bB2, (short)0, acc2, false, false);
        acc3 = __builtin_amdgcn_wmma_f32_16x16x32_bf16(false, a.v, false, bB3, (short)0, acc3, false, false);
      }
      v8f* accs[4] = { &acc0, &acc1, &acc2, &acc3 };
      const int rowbase = NINIT + s * NPS + i0 + rt * 16;
      #pragma unroll
      for (int j = 0; j < 4; ++j) {
        const int col = (q + 4 * j) * 16 + l15;
        const float bias = b2[r * 256 + col];
        #pragma unroll
        for (int v = 0; v < 8; ++v) {
          vectors[(size_t)(rowbase + v + hi * 8) * E + col] = (*accs[j])[v] + bias;
        }
      }
    }

    // ---- device-wide step barrier (8 co-resident WGs)
    __threadfence();
    __syncthreads();
    if (tid == 0) {
      __hip_atomic_fetch_add(bar, 1u, __ATOMIC_RELEASE, __HIP_MEMORY_SCOPE_AGENT);
      const unsigned tgt = (unsigned)(SCAN_BLOCKS * (s + 1));
      while (__hip_atomic_load(bar, __ATOMIC_ACQUIRE, __HIP_MEMORY_SCOPE_AGENT) < tgt) {
        __builtin_amdgcn_s_sleep(2);
      }
    }
    __syncthreads();
    __builtin_amdgcn_fence(__ATOMIC_ACQUIRE, "agent");
  }
}

// ---------------- eval kernel: 2048 WGs x 256 threads, 32 rows each ----------
__global__ __launch_bounds__(256) void eval_kernel(
    const float* __restrict__ vectors,
    const unsigned short* __restrict__ eW1T,
    const float* __restrict__ eb1, const float* __restrict__ eW2,
    const float* __restrict__ eb2,
    const float* __restrict__ pos, const float* __restrict__ neg,
    const float* __restrict__ target,
    const int* __restrict__ mask,
    float* __restrict__ partials)   // [EVAL_BLOCKS][8]
{
  __shared__ unsigned short Xlds[32 * HSTR];
  __shared__ float vlds[32];

  const int tid  = threadIdx.x;
  const int lane = tid & 31;
  const int wave = tid >> 5;
  const int hi   = lane >> 4;
  const int l15  = lane & 15;
  const int rt   = wave & 1;
  const int q    = wave >> 1;

  if (tid < 32) vlds[tid] = 0.f;
  for (int idx = tid; idx < 32 * 256; idx += 256) {
    int row = idx >> 8, k = idx & 255;
    int src = mask[blockIdx.x * 32 + row];
    Xlds[row * HSTR + k] = f2bf(vectors[(size_t)src * E + k]);
  }
  __syncthreads();

  v8f acc0 = {}, acc1 = {}, acc2 = {}, acc3 = {};
  const int arow = (rt * 16 + l15) * HSTR;
  #pragma unroll
  for (int ks = 0; ks < 8; ++ks) {
    const int k0 = ks * 32;
    ABf a;
    a.h[0] = *(const v8bf*)&Xlds[arow + k0 + hi * 16];
    a.h[1] = *(const v8bf*)&Xlds[arow + k0 + hi * 16 + 8];
    const int cb = q * 16 + l15;
    v16bf bB0 = *(const v16bf*)&eW1T[(size_t)(cb)       * 256 + k0 + hi * 16];
    v16bf bB1 = *(const v16bf*)&eW1T[(size_t)(cb +  64) * 256 + k0 + hi * 16];
    v16bf bB2 = *(const v16bf*)&eW1T[(size_t)(cb + 128) * 256 + k0 + hi * 16];
    v16bf bB3 = *(const v16bf*)&eW1T[(size_t)(cb + 192) * 256 + k0 + hi * 16];
    acc0 = __builtin_amdgcn_wmma_f32_16x16x32_bf16(false, a.v, false, bB0, (short)0, acc0, false, false);
    acc1 = __builtin_amdgcn_wmma_f32_16x16x32_bf16(false, a.v, false, bB1, (short)0, acc1, false, false);
    acc2 = __builtin_amdgcn_wmma_f32_16x16x32_bf16(false, a.v, false, bB2, (short)0, acc2, false, false);
    acc3 = __builtin_amdgcn_wmma_f32_16x16x32_bf16(false, a.v, false, bB3, (short)0, acc3, false, false);
  }
  v8f* accs[4] = { &acc0, &acc1, &acc2, &acc3 };
  #pragma unroll
  for (int j = 0; j < 4; ++j) {
    const int col = (q + 4 * j) * 16 + l15;
    const float bias = eb1[col];
    const float w2c  = eW2[col];
    #pragma unroll
    for (int v = 0; v < 8; ++v) {
      float hv = (*accs[j])[v] + bias;
      hv = hv > 0.f ? hv : 0.f;
      atomicAdd(&vlds[rt * 16 + v + hi * 8], hv * w2c);   // ds_add_f32
    }
  }
  __syncthreads();

  if (wave == 0) {
    const int gi = blockIdx.x * 32 + lane;
    const float val = vlds[lane] + eb2[0];
    const float p = pos[gi], nn = neg[gi], t = target[gi];
    float v6[6];
    v6[0] = (p + nn) * t * softplusf(-val);          // A  (scaled by pw later)
    v6[1] = (p + nn) * (1.f - t) * softplusf(val);   // B
    v6[2] = (val >= 0.f) ? p : 0.f;                  // posOK
    v6[3] = (val <  0.f) ? nn : 0.f;                 // negOK
    v6[4] = p;                                       // sum pos
    v6[5] = nn;                                      // sum neg
    #pragma unroll
    for (int o = 16; o > 0; o >>= 1) {
      #pragma unroll
      for (int k = 0; k < 6; ++k) v6[k] += __shfl_down(v6[k], o, 32);
    }
    if (lane == 0) {
      #pragma unroll
      for (int k = 0; k < 6; ++k) partials[(size_t)blockIdx.x * 8 + k] = v6[k];
    }
  }
}

// ---------------- finalize: deterministic-order reduction -> 3 outputs ------
__global__ __launch_bounds__(256) void finalize_kernel(
    const float* __restrict__ partials, float* __restrict__ out)
{
  __shared__ float red[256][6];
  const int tid = threadIdx.x;
  float a[6] = {0.f, 0.f, 0.f, 0.f, 0.f, 0.f};
  for (int b = tid; b < EVAL_BLOCKS; b += 256)
    for (int k = 0; k < 6; ++k) a[k] += partials[(size_t)b * 8 + k];
  for (int k = 0; k < 6; ++k) red[tid][k] = a[k];
  __syncthreads();
  if (tid == 0) {
    float t[6] = {0.f, 0.f, 0.f, 0.f, 0.f, 0.f};
    for (int i = 0; i < 256; ++i)
      for (int k = 0; k < 6; ++k) t[k] += red[i][k];
    const float pw = t[5] / fmaxf(t[4], 1.0f);
    out[0] = pw * t[0] + t[1];   // loss
    out[1] = t[2];               // posOK
    out[2] = t[3];               // negOK
  }
}

// ---------------- launcher ---------------------------------------------------
extern "C" void kernel_launch(void* const* d_in, const int* in_sizes, int n_in,
                              void* d_out, int out_size, void* d_ws, size_t ws_size,
                              hipStream_t stream) {
  (void)in_sizes; (void)n_in; (void)out_size; (void)ws_size;
  const float* init_vecs = (const float*)d_in[0];
  const float* W1    = (const float*)d_in[1];
  const float* b1    = (const float*)d_in[2];
  const float* W2    = (const float*)d_in[3];
  const float* b2    = (const float*)d_in[4];
  const float* eW1   = (const float*)d_in[5];
  const float* eb1   = (const float*)d_in[6];
  const float* eW2   = (const float*)d_in[7];
  const float* eb2   = (const float*)d_in[8];
  const float* pos   = (const float*)d_in[9];
  const float* neg   = (const float*)d_in[10];
  const float* target= (const float*)d_in[11];
  // d_in[12] = ind_steps (rows are contiguous: NINIT + s*NPS + i, recomputed)
  const int* pars    = (const int*)d_in[13];
  const int* rules   = (const int*)d_in[14];
  const int* mask    = (const int*)d_in[15];

  char* ws = (char*)d_ws;
  const size_t VEC_B  = (size_t)TOTAL * E * sizeof(float);            // 135,266,304
  const size_t W1T_B  = (size_t)RULES * 256 * 512 * sizeof(unsigned short);
  const size_t W2T_B  = (size_t)RULES * 256 * 256 * sizeof(unsigned short);
  const size_t EW1T_B = (size_t)256 * 256 * sizeof(unsigned short);
  const size_t PART_B = (size_t)EVAL_BLOCKS * 8 * sizeof(float);

  float*          vectors  = (float*)(ws);
  unsigned short* W1T      = (unsigned short*)(ws + VEC_B);
  unsigned short* W2T      = (unsigned short*)(ws + VEC_B + W1T_B);
  unsigned short* eW1T     = (unsigned short*)(ws + VEC_B + W1T_B + W2T_B);
  float*          partials = (float*)(ws + VEC_B + W1T_B + W2T_B + EW1T_B);
  unsigned int*   bar      = (unsigned int*)(ws + VEC_B + W1T_B + W2T_B + EW1T_B + PART_B);

  const size_t prep_elems = (size_t)RULES*256*512 + (size_t)RULES*256*256
                          + 256*256 + (size_t)NINIT*E;
  const int prep_blocks = (int)((prep_elems + 255) / 256);

  prep_kernel<<<prep_blocks, 256, 0, stream>>>(init_vecs, W1, W2, eW1,
                                               vectors, W1T, W2T, eW1T, bar);
  scan_kernel<<<SCAN_BLOCKS, 256, 0, stream>>>(vectors, W1T, W2T, b1, b2,
                                               pars, rules, bar);
  eval_kernel<<<EVAL_BLOCKS, 256, 0, stream>>>(vectors, eW1T, eb1, eW2, eb2,
                                               pos, neg, target, mask, partials);
  finalize_kernel<<<1, 256, 0, stream>>>(partials, (float*)d_out);
}
